// MultiHeadedSelfAttention_v2_29429115912471
// MI455X (gfx1250) — compile-verified, matrix operand order checked by reading, not run on hardware
//
#include <hip/hip_runtime.h>
#include <hip/hip_bf16.h>
#include <math.h>

// Problem constants (match reference)
#define B_ 64
#define S_ 432
#define D_ 512
#define C_ 128
#define BN_EPS 1e-5f

typedef __attribute__((ext_vector_type(16))) __bf16 v16bf;
typedef __attribute__((ext_vector_type(8)))  float  v8f;
typedef __attribute__((ext_vector_type(4)))  int    v4i;   // builtin's b128 unit
typedef __bf16 bf16_t;

// LDS tile: 64 rows x 32 K-elems, padded to 40 bf16/row (80B) so the 16
// fragment rows hit disjoint bank groups on ds_load_b128 reads.
#define LDS_PAD 40

// CDNA5 async global->LDS path (ASYNCcnt), if the toolchain exposes it.
#if __has_builtin(__builtin_amdgcn_global_load_async_to_lds_b128) && \
    __has_builtin(__builtin_amdgcn_s_wait_asynccnt)
#define HAVE_ASYNC_LDS 1
#else
#define HAVE_ASYNC_LDS 0
#endif

__device__ __forceinline__ void copy_b128(const bf16_t* g, bf16_t* l) {
#if HAVE_ASYNC_LDS
  __builtin_amdgcn_global_load_async_to_lds_b128((v4i*)g, (v4i*)l, 0, 0);
#else
  *(uint4*)l = *(const uint4*)g;
#endif
}

// ---------------------------------------------------------------------------
// LDS fragment read for v_wmma_f32_16x16x32_bf16 (wave32).
// Within a 32-wide K step: elem e<8 -> k = 8*(lane>=16)+e ;
//                          elem e>=8 -> k = 16 + 8*(lane>=16) + (e-8).
// Two contiguous 16B reads per lane (ds_load_b128 x2).
// ---------------------------------------------------------------------------
__device__ __forceinline__ v16bf ldsFrag(const bf16_t* __restrict__ base, int r0) {
  const int l = threadIdx.x & 31;
  const bf16_t* p = base + (r0 + (l & 15)) * LDS_PAD + ((l >> 4) << 3);
  v16bf f;
#pragma unroll
  for (int e = 0; e < 8; ++e) f[e] = p[e];
#pragma unroll
  for (int e = 0; e < 8; ++e) f[e + 8] = p[16 + e];
  return f;
}

__device__ __forceinline__ v8f wmma_bf16(v16bf a, v16bf b, v8f c) {
  return __builtin_amdgcn_wmma_f32_16x16x32_bf16(false, a, false, b,
                                                 (short)0, c, false, false);
}

// ---------------------------------------------------------------------------
// Generic batched WMMA GEMM, LDS-staged and double-buffered.
//   C = alpha_eff * (A @ Bt^T) + bias[col] + beta_diag*(row==col)
//   A  : M x K row-major            (per-batch stride strideA)
//   Bt : N x K row-major (= B^T)    (per-batch stride strideB)
//   alpha_eff = alpha * (bscale ? sqrt(bscale[b]) : 1)
// Macro-tile 64x64 per 128-thread WG (4 waves; each wave owns a 32x32
// quadrant = four 16x16x32 WMMAs per K-step; 8 ds_load_b128 per 4 WMMAs).
// Interior tiles take a wave-uniform unguarded async-copy fast path; edge
// macro-tiles / the K%32 tail zero-fill LDS so WMMA needs no predication.
// ---------------------------------------------------------------------------
__global__ void __launch_bounds__(128) gemm_wmma_bf16(
    const bf16_t* __restrict__ A,  int lda,  long strideA,
    const bf16_t* __restrict__ Bt, int ldbt, long strideB,
    bf16_t* __restrict__ outB, float* __restrict__ outF, int ldo, long strideO,
    int M, int N, int K,
    float alpha, float beta_diag,
    const float* __restrict__ bias,     // per-col bias, may be null
    const float* __restrict__ bscale,   // per-batch sqrt-scale, may be null
    int macroN, int macroPerBatch)
{
  __shared__ __align__(16) bf16_t As[2][64][LDS_PAD];
  __shared__ __align__(16) bf16_t Bs[2][64][LDS_PAD];

  const int wg = blockIdx.x;
  const int b  = wg / macroPerBatch;
  const int t  = wg % macroPerBatch;
  const int m0 = (t / macroN) << 6;
  const int n0 = (t % macroN) << 6;

  const bf16_t* Ab = A  + (long)b * strideA;
  const bf16_t* Bb = Bt + (long)b * strideB;

  const bool fullMN = (m0 + 64 <= M) && (n0 + 64 <= N);

  // copy stage: 2 x 16B chunks per thread per operand per K-step
  auto copyTiles = [&](int buf, int k0) {
    if (fullMN && (k0 + 32 <= K)) {
      // wave-uniform fast path: no per-lane predication
#pragma unroll
      for (int p = 0; p < 2; ++p) {
        const int c   = (p << 7) + threadIdx.x;
        const int row = c >> 2;
        const int cc  = (c & 3) << 3;
        copy_b128(Ab + (long)(m0 + row) * lda  + k0 + cc, &As[buf][row][cc]);
        copy_b128(Bb + (long)(n0 + row) * ldbt + k0 + cc, &Bs[buf][row][cc]);
      }
      return;
    }
#pragma unroll
    for (int p = 0; p < 2; ++p) {
      const int c   = (p << 7) + threadIdx.x;
      const int row = c >> 2;
      const int cc  = (c & 3) << 3;
      const bool kok = (k0 + cc) < K;
      const uint4 z4 = {0u, 0u, 0u, 0u};
      if (kok && (m0 + row) < M)
        copy_b128(Ab + (long)(m0 + row) * lda + k0 + cc, &As[buf][row][cc]);
      else
        *(uint4*)&As[buf][row][cc] = z4;
      if (kok && (n0 + row) < N)
        copy_b128(Bb + (long)(n0 + row) * ldbt + k0 + cc, &Bs[buf][row][cc]);
      else
        *(uint4*)&Bs[buf][row][cc] = z4;
    }
  };

  const int wv = threadIdx.x >> 5;  // wave 0..3
  const int wm = wv & 1;            // 32-row block
  const int wn = wv >> 1;           // 32-col block

  v8f acc[2][2] = {{{}, {}}, {{}, {}}};
  const int nsteps = (K + 31) >> 5;

  copyTiles(0, 0);
  for (int s = 0; s < nsteps; ++s) {
#if HAVE_ASYNC_LDS
    __builtin_amdgcn_s_wait_asynccnt(0);
#endif
    __syncthreads();
    if (s + 1 < nsteps) copyTiles((s + 1) & 1, (s + 1) << 5);
    if (s + 2 < nsteps)  // L2 prefetch hint (global_prefetch_b8)
      __builtin_prefetch(Ab + (long)(m0 + (threadIdx.x >> 2)) * lda +
                             ((s + 2) << 5) + ((threadIdx.x & 3) << 3), 0, 1);
    const int cur = s & 1;
    v16bf a0 = ldsFrag(&As[cur][0][0], (wm << 5));
    v16bf a1 = ldsFrag(&As[cur][0][0], (wm << 5) + 16);
    v16bf b0 = ldsFrag(&Bs[cur][0][0], (wn << 5));
    v16bf b1 = ldsFrag(&Bs[cur][0][0], (wn << 5) + 16);
    acc[0][0] = wmma_bf16(a0, b0, acc[0][0]);
    acc[0][1] = wmma_bf16(a0, b1, acc[0][1]);
    acc[1][0] = wmma_bf16(a1, b0, acc[1][0]);
    acc[1][1] = wmma_bf16(a1, b1, acc[1][1]);
  }

  float a_eff = alpha;
  if (bscale) a_eff *= sqrtf(bscale[b]);
  const int l = threadIdx.x & 31;
  bf16_t* oB = outB ? outB + (long)b * strideO : nullptr;
  float*  oF = outF ? outF + (long)b * strideO : nullptr;
#pragma unroll
  for (int mi = 0; mi < 2; ++mi) {
#pragma unroll
    for (int ni = 0; ni < 2; ++ni) {
      const int col = n0 + (wn << 5) + (ni << 4) + (l & 15);
      if (col >= N) continue;
      const float bcol = bias ? bias[col] : 0.0f;
#pragma unroll
      for (int r = 0; r < 8; ++r) {
        const int row = m0 + (wm << 5) + (mi << 4) + r + ((l >> 4) << 3);
        if (row >= M) continue;
        float v = a_eff * acc[mi][ni][r] + bcol + ((row == col) ? beta_diag : 0.0f);
        const long off = (long)row * ldo + col;
        if (oB) oB[off] = (bf16_t)v;
        if (oF) oF[off] = v;
      }
    }
  }
}

// ---------------------------------------------------------------------------
// x -> bf16(x), bf16(relu(x))
// ---------------------------------------------------------------------------
__global__ void pack_x(const float* __restrict__ x, bf16_t* __restrict__ xh,
                       bf16_t* __restrict__ xrh, long n) {
  long i = (long)blockIdx.x * blockDim.x + threadIdx.x;
  if (i < n) {
    float v = x[i];
    xh[i]  = (bf16_t)v;
    xrh[i] = (bf16_t)fmaxf(v, 0.0f);
  }
}

// ---------------------------------------------------------------------------
// Weights to bf16 in ORIGINAL row-major layout (these are exactly the B^T
// operands the GEMM wants); BN scale folded into Wc, fused bias for conv+BN.
// ---------------------------------------------------------------------------
__global__ void pack_weights(const float* __restrict__ Wv,
                             const float* __restrict__ Wc,
                             const float* __restrict__ bc,
                             const float* __restrict__ gamma,
                             const float* __restrict__ beta,
                             const float* __restrict__ mean,
                             const float* __restrict__ var,
                             bf16_t* __restrict__ Wvb,
                             bf16_t* __restrict__ Wcb,
                             float* __restrict__ biasc) {
  int i = blockIdx.x * blockDim.x + threadIdx.x;
  if (i < D_ * D_) Wvb[i] = (bf16_t)Wv[i];
  if (i < C_ * D_) {
    int c = i / D_;
    float sc = gamma[c] * rsqrtf(var[c] + BN_EPS);
    Wcb[i] = (bf16_t)(Wc[i] * sc);
  }
  if (i < C_) {
    float sc = gamma[i] * rsqrtf(var[i] + BN_EPS);
    biasc[i] = (bc[i] - mean[i]) * sc + beta[i];
  }
}

// ---------------------------------------------------------------------------
// Per-(b,s) mean over C; fc = feat - mean and feat, both bf16 row-major SxC.
// (feat row-major IS the cov-pool B^T operand.) One wave per row.
// ---------------------------------------------------------------------------
__global__ void __launch_bounds__(256) row_center(const float* __restrict__ featf,
                                                  bf16_t* __restrict__ fcb,
                                                  bf16_t* __restrict__ featb) {
  const int wave = (int)((blockIdx.x * blockDim.x + threadIdx.x) >> 5);
  const int l = threadIdx.x & 31;
  const float* row = featf + (long)wave * C_;
  float v[4], sum = 0.0f;
#pragma unroll
  for (int j = 0; j < 4; ++j) { v[j] = row[l + 32 * j]; sum += v[j]; }
#pragma unroll
  for (int off = 16; off > 0; off >>= 1) sum += __shfl_xor(sum, off, 32);
  const float mean = sum * (1.0f / C_);
  bf16_t* fr = fcb   + (long)wave * C_;
  bf16_t* fb = featb + (long)wave * C_;
#pragma unroll
  for (int j = 0; j < 4; ++j) {
    int c = l + 32 * j;
    fr[c] = (bf16_t)(v[j] - mean);
    fb[c] = (bf16_t)v[j];
  }
}

// ---------------------------------------------------------------------------
// 32x32 LDS-tiled transpose per batch: vh (S x D) -> vT (D x S), bf16.
// ---------------------------------------------------------------------------
__global__ void __launch_bounds__(256) transpose_v(const bf16_t* __restrict__ in,
                                                   bf16_t* __restrict__ outT) {
  __shared__ bf16_t tile[32][33];
  const int b = blockIdx.z;
  const int s0 = blockIdx.y << 5, d0 = blockIdx.x << 5;
  const bf16_t* ib = in  + (long)b * S_ * D_;
  bf16_t* ob       = outT + (long)b * D_ * S_;
  for (int j = threadIdx.y; j < 32; j += 8) {
    int s = s0 + j, d = d0 + threadIdx.x;
    tile[j][threadIdx.x] = (s < S_) ? ib[(long)s * D_ + d] : (bf16_t)0.0f;
  }
  __syncthreads();
  for (int j = threadIdx.y; j < 32; j += 8) {
    int d = d0 + j, s = s0 + threadIdx.x;
    if (s < S_) ob[(long)d * S_ + s] = tile[threadIdx.x][j];
  }
}

// ---------------------------------------------------------------------------
// normA[b] = trace(cov_b)
// ---------------------------------------------------------------------------
__global__ void __launch_bounds__(256) trace_k(const float* __restrict__ covf,
                                               float* __restrict__ normA) {
  __shared__ float red[256];
  const int b = blockIdx.x;
  const float* Cv = covf + (long)b * S_ * S_;
  float s = 0.0f;
  for (int i = threadIdx.x; i < S_; i += 256) s += Cv[(long)i * S_ + i];
  red[threadIdx.x] = s;
  __syncthreads();
  for (int o = 128; o > 0; o >>= 1) {
    if (threadIdx.x < o) red[threadIdx.x] += red[threadIdx.x + o];
    __syncthreads();
  }
  if (threadIdx.x == 0) normA[b] = red[0];
}

// ---------------------------------------------------------------------------
// Ah = cov/normA ; ZY0 = 1.5*I - 0.5*Ah   (bf16 NS operands)
// ---------------------------------------------------------------------------
__global__ void ns_init(const float* __restrict__ covf,
                        const float* __restrict__ normA,
                        bf16_t* __restrict__ Ahb, bf16_t* __restrict__ ZYb,
                        long total) {
  long i = (long)blockIdx.x * blockDim.x + threadIdx.x;
  if (i >= total) return;
  const long nn = (long)S_ * S_;
  int b = (int)(i / nn);
  long rc = i % nn;
  int r = (int)(rc / S_), c = (int)(rc % S_);
  float a = covf[i] / normA[b];
  Ahb[i] = (bf16_t)a;
  ZYb[i] = (bf16_t)(((r == c) ? 1.5f : 0.0f) - 0.5f * a);
}

// ---------------------------------------------------------------------------
// Row softmax over S=432 cols; one 256-thread block per row; bf16 out.
// ---------------------------------------------------------------------------
__global__ void __launch_bounds__(256) softmax_k(const float* __restrict__ scores,
                                                 bf16_t* __restrict__ attnb) {
  __shared__ float red[256];
  const long row = blockIdx.x;
  const float* p = scores + row * (long)S_;
  bf16_t* o = attnb + row * (long)S_;
  const int t = threadIdx.x;
  float v0 = (t < S_) ? p[t] : -1e30f;
  float v1 = (t + 256 < S_) ? p[t + 256] : -1e30f;
  red[t] = fmaxf(v0, v1);
  __syncthreads();
  for (int w = 128; w > 0; w >>= 1) {
    if (t < w) red[t] = fmaxf(red[t], red[t + w]);
    __syncthreads();
  }
  const float mx = red[0];
  __syncthreads();
  float e0 = (t < S_) ? __expf(v0 - mx) : 0.0f;
  float e1 = (t + 256 < S_) ? __expf(v1 - mx) : 0.0f;
  red[t] = e0 + e1;
  __syncthreads();
  for (int w = 128; w > 0; w >>= 1) {
    if (t < w) red[t] += red[t + w];
    __syncthreads();
  }
  const float inv = 1.0f / red[0];
  if (t < S_) o[t] = (bf16_t)(e0 * inv);
  if (t + 256 < S_) o[t + 256] = (bf16_t)(e1 * inv);
}

// ---------------------------------------------------------------------------
// Host orchestration
// ---------------------------------------------------------------------------
extern "C" void kernel_launch(void* const* d_in, const int* in_sizes, int n_in,
                              void* d_out, int out_size, void* d_ws, size_t ws_size,
                              hipStream_t stream) {
  const float* x     = (const float*)d_in[0];
  // d_in[1]: mask (all-ones; unused by the reference math) -> ignored
  const float* Wv    = (const float*)d_in[2];
  const float* bv    = (const float*)d_in[3];
  const float* Wc    = (const float*)d_in[4];
  const float* bc    = (const float*)d_in[5];
  const float* gamma = (const float*)d_in[6];
  const float* beta  = (const float*)d_in[7];
  const float* mean  = (const float*)d_in[8];
  const float* var   = (const float*)d_in[9];
  float* out = (float*)d_out;

  const long BSD = (long)B_ * S_ * D_;
  const long BSC = (long)B_ * S_ * C_;
  const long BSS = (long)B_ * S_ * S_;
  const long SxS = (long)S_ * S_;

  char* w = (char*)d_ws;
  auto alloc = [&](size_t bytes) -> char* {
    char* p = w;
    w += (bytes + 255) & ~(size_t)255;
    return p;
  };
  bf16_t* xh    = (bf16_t*)alloc(BSD * sizeof(bf16_t));
  bf16_t* xrh   = (bf16_t*)alloc(BSD * sizeof(bf16_t));
  bf16_t* Wvb   = (bf16_t*)alloc((size_t)D_ * D_ * sizeof(bf16_t));
  bf16_t* Wcb   = (bf16_t*)alloc((size_t)C_ * D_ * sizeof(bf16_t));
  float*  biasc = (float*) alloc(C_ * sizeof(float));
  bf16_t* vh    = (bf16_t*)alloc(BSD * sizeof(bf16_t));
  bf16_t* vT    = (bf16_t*)alloc(BSD * sizeof(bf16_t));
  float*  featf = (float*) alloc(BSC * sizeof(float));
  bf16_t* fcb   = (bf16_t*)alloc(BSC * sizeof(bf16_t));
  bf16_t* featb = (bf16_t*)alloc(BSC * sizeof(bf16_t));
  float*  covf  = (float*) alloc(BSS * sizeof(float));   // reused as scores
  float*  normA = (float*) alloc(B_ * sizeof(float));
  bf16_t* Nbuf[5];
  for (int i = 0; i < 5; ++i) Nbuf[i] = (bf16_t*)alloc(BSS * sizeof(bf16_t));
  bf16_t* attnb = (bf16_t*)alloc(BSS * sizeof(bf16_t));
  float*  scores = covf;  // cov fully consumed (trace + ns_init) before reuse

  auto launch_gemm = [&](const bf16_t* A, int lda, long sA,
                         const bf16_t* Bt, int ldbt, long sB,
                         bf16_t* oB, float* oF, int ldo, long sO,
                         int M, int N, int K, float alpha, float beta_d,
                         const float* bias, const float* bscale, int nbatch) {
    int mM = (M + 63) >> 6, mN = (N + 63) >> 6;
    gemm_wmma_bf16<<<nbatch * mM * mN, 128, 0, stream>>>(
        A, lda, sA, Bt, ldbt, sB, oB, oF, ldo, sO,
        M, N, K, alpha, beta_d, bias, bscale, mN, mM * mN);
  };

  // 1) activations -> bf16 (raw + relu)
  pack_x<<<(int)((BSD + 255) / 256), 256, 0, stream>>>(x, xh, xrh, BSD);

  // 2) weights -> bf16 (B^T layout == original row-major), BN folded
  pack_weights<<<(D_ * D_ + 255) / 256, 256, 0, stream>>>(
      Wv, Wc, bc, gamma, beta, mean, var, Wvb, Wcb, biasc);

  // 3) v = x @ Wv^T + bv   (M=27648, N=512, K=512) -> bf16
  launch_gemm(xh, D_, 0, Wvb, D_, 0, vh, nullptr, D_, 0,
              B_ * S_, D_, D_, 1.0f, 0.0f, bv, nullptr, 1);

  // 4) vT per batch (B^T operand of the final attn@v GEMM)
  transpose_v<<<dim3(D_ / 32, (S_ + 31) / 32, B_), dim3(32, 8), 0, stream>>>(vh, vT);

  // 5) feat = BN(relu(x) @ Wc^T + bc)   (M=27648, N=128, K=512) -> f32
  launch_gemm(xrh, D_, 0, Wcb, D_, 0, nullptr, featf, C_, 0,
              B_ * S_, C_, D_, 1.0f, 0.0f, biasc, nullptr, 1);

  // 6) center over C -> fc (bf16) and feat (bf16), both SxC row-major
  row_center<<<(B_ * S_) / 8, 256, 0, stream>>>(featf, fcb, featb);

  // 7) cov = fc @ feat^T / C  per batch (M=N=432, K=128) -> f32
  launch_gemm(fcb, C_, (long)S_ * C_, featb, C_, (long)S_ * C_,
              nullptr, covf, S_, SxS, S_, S_, C_, 1.0f / C_, 0.0f,
              nullptr, nullptr, B_);

  // 8) normA = trace(cov)
  trace_k<<<B_, 256, 0, stream>>>(covf, normA);

  // 9) Ah = cov/normA, ZY0 = 1.5I - 0.5 Ah
  ns_init<<<(int)((BSS + 255) / 256), 256, 0, stream>>>(covf, normA,
                                                        Nbuf[0], Nbuf[1], BSS);

  // NS GEMMs: symmetric operands => B^T == B (same row-major buffer)
  auto gemmNS = [&](const bf16_t* A, const bf16_t* Bm, bf16_t* oB, float* oF,
                    float alpha, float beta_d, const float* bscale) {
    launch_gemm(A, S_, SxS, Bm, S_, SxS, oB, oF, S_, SxS,
                S_, S_, S_, alpha, beta_d, nullptr, bscale, B_);
  };

  // 10) Newton-Schulz: Y1 = Ah @ ZY0 ; Z1 = ZY0
  gemmNS(Nbuf[0], Nbuf[1], Nbuf[2], nullptr, 1.0f, 0.0f, nullptr);
  bf16_t* Y = Nbuf[2];
  bf16_t* Z = Nbuf[1];
  bf16_t* freeb[3] = {Nbuf[0], Nbuf[3], Nbuf[4]};
  for (int it = 0; it < 3; ++it) {  // ITERN-2
    bf16_t* zy = freeb[0]; bf16_t* yn = freeb[1]; bf16_t* zn = freeb[2];
    gemmNS(Z, Y, zy, nullptr, -0.5f, 1.5f, nullptr);  // ZY = 1.5I - 0.5 Z@Y
    gemmNS(Y, zy, yn, nullptr, 1.0f, 0.0f, nullptr);  // Y' = Y@ZY
    gemmNS(zy, Z, zn, nullptr, 1.0f, 0.0f, nullptr);  // Z' = ZY@Z
    freeb[0] = Y; freeb[1] = Z; freeb[2] = zy;
    Y = yn; Z = zn;
  }
  // final NS step: scores = (Y @ (1.5I - 0.5 Z@Y)) * sqrt(normA) / 16
  bf16_t* zyf = freeb[0];
  gemmNS(Z, Y, zyf, nullptr, -0.5f, 1.5f, nullptr);
  gemmNS(Y, zyf, nullptr, scores, 1.0f / 16.0f, 0.0f, normA);

  // 11) softmax rows -> bf16 attn
  softmax_k<<<B_ * S_, 256, 0, stream>>>(scores, attnb);

  // 12) out = attn @ v  per batch (M=432, N=512, K=432) -> f32 d_out
  launch_gemm(attnb, S_, SxS, vT, S_, (long)D_ * S_,
              nullptr, out, D_, (long)S_ * D_,
              S_, D_, S_, 1.0f, 0.0f, nullptr, nullptr, B_);

  (void)in_sizes; (void)n_in; (void)out_size; (void)ws_size;
}
